// DecoderRNN_20504173871200
// MI455X (gfx1250) — matrix operand order repaired and checked by reading, not compiled, not run on hardware
//
#include <hip/hip_runtime.h>
#include <hip/hip_bf16.h>

// DecoderRNN (show-attend-tell style) for MI455X / gfx1250.
// bf16 weights pre-transposed to [N,K]; all GEMMs run on v_wmma_f32_16x16x32_bf16.
// Register blocking: each wave owns a 64x16 output strip (4 M-tiles sharing one
// B fragment) -> 4 WMMAs per B load, halving load instructions per MAC and
// cutting B traffic 4x for the dominant 64x32000x1024 vocab projection.
// bf16 weights (~106 MB) stay L2-resident (192 MB): steady state is XDL-bound.

#define B_   64
#define N_   49
#define ENC_ 2048
#define DEC_ 1024
#define EMB_ 512
#define ATT_ 512
#define V_   32000
#define T_   25
#define TS_  24                // T-1 decode steps
#define KCOMB_ 3584            // EMB + ENC + DEC
#define G4_  4096              // 4*DEC
#define BN_  (B_ * N_)         // 3136 rows for encoder projection

typedef __bf16 bf16_t;
typedef __attribute__((ext_vector_type(16))) __bf16 v16bf;
typedef __attribute__((ext_vector_type(8)))  __bf16 v8bf;
typedef __attribute__((ext_vector_type(8)))  float  v8f;

#define CDIV(a, b) (((a) + (b) - 1) / (b))

__device__ __forceinline__ bf16_t f2bf(float f) {
  unsigned int u = __float_as_uint(f);
  unsigned int r = (u + 0x7FFFu + ((u >> 16) & 1u)) >> 16;   // RNE
  unsigned short s = (unsigned short)r;
  return __builtin_bit_cast(bf16_t, s);
}

__device__ __forceinline__ float sigmoidf_(float x) {
  return 1.0f / (1.0f + expf(-x));
}

// ---------------------------------------------------------------------------
// bf16 WMMA GEMM:  C[M,Nout] = A[M,K] @ Bw[Nout,K]^T + bias[Nout]
// Each wave computes a 64x16 strip (4 stacked 16x16 tiles) so one B fragment
// feeds 4 WMMAs. 8 waves per block tile along N.
// Requires: M multiple of 64 (grid.y = M/64), Nout multiple of 16,
//           K multiple of 32, lda/ldb multiples of 16 elements.
// ---------------------------------------------------------------------------
__global__ __launch_bounds__(256)
void k_gemm_bf16(const bf16_t* __restrict__ A, int lda,
                 const bf16_t* __restrict__ Bw, int ldb,
                 const float* __restrict__ bias,
                 float* __restrict__ C, long long ldc,
                 int Nout, int K)
{
  const int wave = threadIdx.x >> 5;
  const int lane = threadIdx.x & 31;
  const int n0 = (blockIdx.x * 8 + wave) * 16;
  if (n0 >= Nout) return;
  const int m0 = blockIdx.y * 64;

  const int half = lane >> 4;          // 0: lanes 0-15, 1: lanes 16-31
  const int l16  = lane & 15;
  // A layout (16-bit, 16x32): half0 holds K 0..15, half1 K 16..31 (matches B)
  const int aoff0 = half * 16;
  const int aoff1 = half * 16 + 8;
  const bf16_t* Ap0 = A + (long long)(m0 + l16) * lda;
  const bf16_t* Ap1 = Ap0 + 16ll * lda;
  const bf16_t* Ap2 = Ap0 + 32ll * lda;
  const bf16_t* Ap3 = Ap0 + 48ll * lda;
  // B layout (16-bit, 32x16): column = n0+l16 = row of Bw;
  // lanes 0-15 hold K 0..15, lanes 16-31 hold K 16..31 (contiguous 32 B).
  const bf16_t* Bptr = Bw + (long long)(n0 + l16) * ldb + half * 16;

  v8f acc0 = {}, acc1 = {}, acc2 = {}, acc3 = {};
  for (int k = 0; k < K; k += 32) {
    v16bf b = *(const v16bf*)(Bptr + k);
    union { v16bf v; v8bf h[2]; } a0, a1, a2, a3;
    a0.h[0] = *(const v8bf*)(Ap0 + k + aoff0);
    a0.h[1] = *(const v8bf*)(Ap0 + k + aoff1);
    a1.h[0] = *(const v8bf*)(Ap1 + k + aoff0);
    a1.h[1] = *(const v8bf*)(Ap1 + k + aoff1);
    a2.h[0] = *(const v8bf*)(Ap2 + k + aoff0);
    a2.h[1] = *(const v8bf*)(Ap2 + k + aoff1);
    a3.h[0] = *(const v8bf*)(Ap3 + k + aoff0);
    a3.h[1] = *(const v8bf*)(Ap3 + k + aoff1);
    acc0 = __builtin_amdgcn_wmma_f32_16x16x32_bf16(
        false, a0.v, false, b, (short)0, acc0, false, false);
    acc1 = __builtin_amdgcn_wmma_f32_16x16x32_bf16(
        false, a1.v, false, b, (short)0, acc1, false, false);
    acc2 = __builtin_amdgcn_wmma_f32_16x16x32_bf16(
        false, a2.v, false, b, (short)0, acc2, false, false);
    acc3 = __builtin_amdgcn_wmma_f32_16x16x32_bf16(
        false, a3.v, false, b, (short)0, acc3, false, false);
  }

  const float bv = bias ? bias[n0 + l16] : 0.0f;
  const long long col = n0 + l16;
  // C/D layout: lane group half -> rows mb+8*half, VGPR r -> row +r
#define STORE_TILE(ACC, MT)                                            \
  {                                                                    \
    const int mb = m0 + (MT) * 16 + half * 8;                          \
    _Pragma("unroll") for (int r = 0; r < 8; ++r)                      \
        C[(long long)(mb + r) * ldc + col] = (ACC)[r] + bv;            \
  }
  STORE_TILE(acc0, 0)
  STORE_TILE(acc1, 1)
  STORE_TILE(acc2, 2)
  STORE_TILE(acc3, 3)
#undef STORE_TILE
}

// ---------------------------------------------------------------------------
// One-time conversion / packing kernels
// ---------------------------------------------------------------------------
__global__ void k_f2bf(const float* __restrict__ src, bf16_t* __restrict__ dst,
                       long long n) {
  long long i = (long long)blockIdx.x * 256 + threadIdx.x;
  if (i < n) dst[i] = f2bf(src[i]);
}

// src [R,C] row-major -> dst [C,R] bf16 (dst[c*R + r] = src[r*C + c])
__global__ void k_transpose_bf(const float* __restrict__ src,
                               bf16_t* __restrict__ dst, int R, int C) {
  long long i = (long long)blockIdx.x * 256 + threadIdx.x;
  if (i >= (long long)R * C) return;
  int c = (int)(i / R);
  int r = (int)(i % R);
  dst[i] = f2bf(src[(long long)r * C + c]);
}

// pack src[rows, csrc] into dst rows with leading dim ldd at column offset coff
__global__ void k_pack_cols(const float* __restrict__ src,
                            bf16_t* __restrict__ dst, int rows, int csrc,
                            int ldd, int coff) {
  long long i = (long long)blockIdx.x * 256 + threadIdx.x;
  if (i >= (long long)rows * csrc) return;
  int r = (int)(i / csrc);
  int c = (int)(i % csrc);
  dst[(long long)r * ldd + coff + c] = f2bf(src[i]);
}

__global__ void k_add2(const float* __restrict__ a, const float* __restrict__ b,
                       float* __restrict__ o, int n) {
  int i = blockIdx.x * 256 + threadIdx.x;
  if (i < n) o[i] = a[i] + b[i];
}

// mean over N of features -> bf16 [B, ENC]
__global__ __launch_bounds__(256)
void k_mean(const float* __restrict__ features, bf16_t* __restrict__ meanA) {
  const int b = blockIdx.x;
  const float* fb = features + (long long)b * N_ * ENC_;
  for (int e = threadIdx.x; e < ENC_; e += 256) {
    float s = 0.0f;
    for (int n = 0; n < N_; ++n) s += fb[n * ENC_ + e];
    meanA[b * ENC_ + e] = f2bf(s * (1.0f / (float)N_));
  }
}

// gather captions[:, 0:T-1] embeddings -> bf16 [TS, B, EMB]
__global__ void k_embeds(const int* __restrict__ captions,
                         const float* __restrict__ emb,
                         bf16_t* __restrict__ out) {
  long long i = (long long)blockIdx.x * 256 + threadIdx.x;
  if (i >= (long long)TS_ * B_ * EMB_) return;
  int t = (int)(i / (B_ * EMB_));
  int r = (int)(i % (B_ * EMB_));
  int b = r >> 9;                       // / EMB_
  int e = r & (EMB_ - 1);
  int tok = captions[b * T_ + t];
  out[i] = f2bf(emb[(long long)tok * EMB_ + e]);
}

// h0 f32 [B, DEC] -> bf16 into Acomb h-slot
__global__ void k_h0_to_acomb(const float* __restrict__ h0f,
                              bf16_t* __restrict__ Acomb) {
  int i = blockIdx.x * 256 + threadIdx.x;
  if (i >= B_ * DEC_) return;
  int b = i >> 10;
  int j = i & (DEC_ - 1);
  Acomb[(long long)b * KCOMB_ + EMB_ + ENC_ + j] = f2bf(h0f[i]);
}

// ---------------------------------------------------------------------------
// Per-step kernels
// ---------------------------------------------------------------------------
// One block per batch element: scores -> softmax -> alpha out, context -> Acomb,
// and copy this step's embedding into the Acomb x-slot.
__global__ __launch_bounds__(256)
void k_attention(const float* __restrict__ encp,    // [B*N, ATT]
                 const float* __restrict__ hproj,   // [B, ATT] (incl. Wd_b)
                 const float* __restrict__ va,      // [ATT]
                 const float* __restrict__ va_b,    // [1]
                 const float* __restrict__ features,// [B, N, ENC] f32
                 const bf16_t* __restrict__ xemb,   // [B, EMB] this step
                 bf16_t* __restrict__ Acomb,        // [B, KCOMB]
                 float* __restrict__ attn)          // + b*(TS*N) + n
{
  const int b = blockIdx.x;
  const int tid = threadIdx.x;
  __shared__ float s_h[ATT_];
  __shared__ float s_sc[64];

  for (int i = tid; i < ATT_; i += 256) s_h[i] = hproj[b * ATT_ + i];
  __syncthreads();

  if (tid < N_) {
    const float* ep = encp + (long long)(b * N_ + tid) * ATT_;
    float s = va_b[0];
    for (int a = 0; a < ATT_; ++a) s = fmaf(va[a], tanhf(ep[a] + s_h[a]), s);
    s_sc[tid] = s;
  }
  __syncthreads();

  if (tid == 0) {
    float mx = s_sc[0];
    for (int n = 1; n < N_; ++n) mx = fmaxf(mx, s_sc[n]);
    float sum = 0.0f;
    for (int n = 0; n < N_; ++n) { float e = expf(s_sc[n] - mx); s_sc[n] = e; sum += e; }
    float inv = 1.0f / sum;
    for (int n = 0; n < N_; ++n) s_sc[n] *= inv;
  }
  __syncthreads();

  if (tid < N_) attn[(long long)b * (TS_ * N_) + tid] = s_sc[tid];

  const float* fb = features + (long long)b * N_ * ENC_;
  bf16_t* arow = Acomb + (long long)b * KCOMB_;
  for (int e = tid; e < ENC_; e += 256) {
    float ctx = 0.0f;
    for (int n = 0; n < N_; ++n) ctx = fmaf(s_sc[n], fb[n * ENC_ + e], ctx);
    arow[EMB_ + e] = f2bf(ctx);
  }
  for (int e = tid; e < EMB_; e += 256) arow[e] = xemb[b * EMB_ + e];
}

// LSTM pointwise: gates [B, 4*DEC] (biases pre-added), c state f32 updated,
// h_new written bf16 into the Acomb h-slot (used by this step's vocab GEMM
// and next step's hproj GEMM).
__global__ void k_lstm(const float* __restrict__ gates,
                       float* __restrict__ c, bf16_t* __restrict__ Acomb) {
  int i = blockIdx.x * 256 + threadIdx.x;
  if (i >= B_ * DEC_) return;
  int b = i >> 10;
  int j = i & (DEC_ - 1);
  const float* g = gates + (long long)b * G4_;
  float ig = sigmoidf_(g[j]);
  float fg = sigmoidf_(g[DEC_ + j]);
  float gg = tanhf(g[2 * DEC_ + j]);
  float og = sigmoidf_(g[3 * DEC_ + j]);
  float cn = fg * c[i] + ig * gg;
  c[i] = cn;
  Acomb[(long long)b * KCOMB_ + EMB_ + ENC_ + j] = f2bf(og * tanhf(cn));
}

// ---------------------------------------------------------------------------
extern "C" void kernel_launch(void* const* d_in, const int* in_sizes, int n_in,
                              void* d_out, int out_size, void* d_ws, size_t ws_size,
                              hipStream_t stream)
{
  (void)in_sizes; (void)n_in; (void)out_size; (void)ws_size;

  const float* features = (const float*)d_in[0];
  const int*   captions = (const int*)d_in[1];
  const float* emb   = (const float*)d_in[2];
  const float* Ue    = (const float*)d_in[3];
  const float* Ue_b  = (const float*)d_in[4];
  const float* Wd    = (const float*)d_in[5];
  const float* Wd_b  = (const float*)d_in[6];
  const float* va    = (const float*)d_in[7];
  const float* va_b  = (const float*)d_in[8];
  const float* Wh    = (const float*)d_in[9];
  const float* bh    = (const float*)d_in[10];
  const float* Wc    = (const float*)d_in[11];
  const float* bc    = (const float*)d_in[12];
  const float* Wih   = (const float*)d_in[13];
  const float* Whh   = (const float*)d_in[14];
  const float* bih   = (const float*)d_in[15];
  const float* bhh   = (const float*)d_in[16];
  const float* Wfc   = (const float*)d_in[17];
  const float* bfc   = (const float*)d_in[18];

  // ---- workspace layout (all chunk sizes are multiples of 256 B) ----
  char* ws = (char*)d_ws;
  size_t off = 0;
  auto take = [&](size_t bytes) { char* p = ws + off; off += bytes; return p; };

  bf16_t* fA    = (bf16_t*)take((size_t)BN_ * ENC_ * 2);        // features bf16
  bf16_t* UeT   = (bf16_t*)take((size_t)ATT_ * ENC_ * 2);       // [ATT, ENC]
  bf16_t* WdT   = (bf16_t*)take((size_t)ATT_ * DEC_ * 2);       // [ATT, DEC]
  bf16_t* WhT   = (bf16_t*)take((size_t)DEC_ * ENC_ * 2);       // [DEC, ENC]
  bf16_t* WcT   = (bf16_t*)take((size_t)DEC_ * ENC_ * 2);       // [DEC, ENC]
  bf16_t* Wcomb = (bf16_t*)take((size_t)G4_ * KCOMB_ * 2);      // [4H, Kcomb]
  bf16_t* WfcT  = (bf16_t*)take((size_t)V_ * DEC_ * 2);         // [V, DEC]
  float*  bcomb = (float*) take((size_t)G4_ * 4);
  bf16_t* meanA = (bf16_t*)take((size_t)B_ * ENC_ * 2);
  float*  encp  = (float*) take((size_t)BN_ * ATT_ * 4);        // enc projection
  bf16_t* embB  = (bf16_t*)take((size_t)TS_ * B_ * EMB_ * 2);
  bf16_t* Acomb = (bf16_t*)take((size_t)B_ * KCOMB_ * 2);       // [x|ctx|h]
  float*  hproj = (float*) take((size_t)B_ * ATT_ * 4);
  float*  gates = (float*) take((size_t)B_ * G4_ * 4);
  float*  cst   = (float*) take((size_t)B_ * DEC_ * 4);
  float*  h0f   = (float*) take((size_t)B_ * DEC_ * 4);
  // total ~130 MB

  // ---- one-time conversions ----
  k_f2bf<<<CDIV((long long)BN_ * ENC_, 256), 256, 0, stream>>>(
      features, fA, (long long)BN_ * ENC_);
  k_transpose_bf<<<CDIV(ENC_ * ATT_, 256), 256, 0, stream>>>(Ue, UeT, ENC_, ATT_);
  k_transpose_bf<<<CDIV(DEC_ * ATT_, 256), 256, 0, stream>>>(Wd, WdT, DEC_, ATT_);
  k_transpose_bf<<<CDIV(ENC_ * DEC_, 256), 256, 0, stream>>>(Wh, WhT, ENC_, DEC_);
  k_transpose_bf<<<CDIV(ENC_ * DEC_, 256), 256, 0, stream>>>(Wc, WcT, ENC_, DEC_);
  k_transpose_bf<<<CDIV((long long)DEC_ * V_, 256), 256, 0, stream>>>(
      Wfc, WfcT, DEC_, V_);
  k_pack_cols<<<CDIV(G4_ * (EMB_ + ENC_), 256), 256, 0, stream>>>(
      Wih, Wcomb, G4_, EMB_ + ENC_, KCOMB_, 0);
  k_pack_cols<<<CDIV(G4_ * DEC_, 256), 256, 0, stream>>>(
      Whh, Wcomb, G4_, DEC_, KCOMB_, EMB_ + ENC_);
  k_add2<<<CDIV(G4_, 256), 256, 0, stream>>>(bih, bhh, bcomb, G4_);
  k_mean<<<B_, 256, 0, stream>>>(features, meanA);
  k_embeds<<<CDIV(TS_ * B_ * EMB_, 256), 256, 0, stream>>>(captions, emb, embB);

  // ---- h0 / c0 from mean-pooled features (M=64 -> grid.y=1) ----
  k_gemm_bf16<<<dim3(DEC_ / 16 / 8, 1), 256, 0, stream>>>(
      meanA, ENC_, WhT, ENC_, bh, h0f, DEC_, DEC_, ENC_);
  k_gemm_bf16<<<dim3(DEC_ / 16 / 8, 1), 256, 0, stream>>>(
      meanA, ENC_, WcT, ENC_, bc, cst, DEC_, DEC_, ENC_);
  k_h0_to_acomb<<<CDIV(B_ * DEC_, 256), 256, 0, stream>>>(h0f, Acomb);

  // ---- time-invariant encoder projection: [3136,2048]@[2048,512] ----
  k_gemm_bf16<<<dim3(ATT_ / 16 / 8, BN_ / 64), 256, 0, stream>>>(
      fA, ENC_, UeT, ENC_, Ue_b, encp, ATT_, ATT_, ENC_);

  float* preds = (float*)d_out;                       // [B, TS, V]
  float* attnB = preds + (long long)B_ * TS_ * V_;    // [B, TS, N]
  bf16_t* Ah = Acomb + (EMB_ + ENC_);                 // h-slot view, lda=KCOMB

  for (int t = 0; t < TS_; ++t) {
    // 1) hproj = h @ Wd + Wd_b
    k_gemm_bf16<<<dim3(ATT_ / 16 / 8, 1), 256, 0, stream>>>(
        Ah, KCOMB_, WdT, DEC_, Wd_b, hproj, ATT_, ATT_, DEC_);
    // 2) attention + context + x_t staging
    k_attention<<<B_, 256, 0, stream>>>(
        encp, hproj, va, va_b, features,
        embB + (long long)t * B_ * EMB_, Acomb, attnB + (long long)t * N_);
    // 3) gates = [x|ctx|h] @ [Wih|Whh]^T + (bih+bhh)
    k_gemm_bf16<<<dim3(G4_ / 16 / 8, 1), 256, 0, stream>>>(
        Acomb, KCOMB_, Wcomb, KCOMB_, bcomb, gates, G4_, G4_, KCOMB_);
    // 4) LSTM pointwise -> c (f32), h_new (bf16 into Acomb h-slot)
    k_lstm<<<CDIV(B_ * DEC_, 256), 256, 0, stream>>>(gates, cst, Acomb);
    // 5) preds[:, t, :] = h_new @ Wfc + bfc  (straight into d_out)
    k_gemm_bf16<<<dim3(V_ / 16 / 8, 1), 256, 0, stream>>>(
        Ah, KCOMB_, WfcT, DEC_, bfc, preds + (long long)t * V_,
        (long long)TS_ * V_, V_, DEC_);
  }
}
